// RNNLayer_81647328297080
// MI455X (gfx1250) — compile-verified, matrix operand order checked by reading, not run on hardware
//
#include <hip/hip_runtime.h>
#include <hip/hip_bf16.h>

#define BATCH  64
#define SEQ    2048
#define IN_DIM 512
#define HID    512

#define LDS_STRIDE 520   // 512 + 8 bf16 pad; 1040 B row = 16B aligned
#define MT 64            // rows per block in the big GEMM

typedef __bf16        v16bf __attribute__((ext_vector_type(16)));
typedef __bf16        v4bf  __attribute__((ext_vector_type(4)));
typedef float         v8f   __attribute__((ext_vector_type(8)));
typedef unsigned int  v4u   __attribute__((ext_vector_type(4)));

// One WMMA 16-bit operand fragment: 16 bf16 = two 16-byte chunks per lane.
// ISA 7.12.2 16-bit 16x32 layout: chunk0 = k[8*hf .. 8*hf+7],
// chunk1 = k[16+8*hf .. 16+8*hf+7] within each 32-wide k-block.
union Frag {
    v16bf v;
    v4u   q[2];
};

// ---------------------------------------------------------------------------
// Kernel 1: fp32 -> bf16 weight conversion into workspace
// ---------------------------------------------------------------------------
__global__ void rnn_cvt_weights(const float* __restrict__ wih,
                                const float* __restrict__ whh,
                                __bf16* __restrict__ wih_bf,
                                __bf16* __restrict__ whh_bf) {
    int i = blockIdx.x * blockDim.x + threadIdx.x;   // i indexes float4 groups
    if (i < HID * IN_DIM / 4) {
        float4 a = ((const float4*)wih)[i];
        float4 b = ((const float4*)whh)[i];
        v4bf va = { (__bf16)a.x, (__bf16)a.y, (__bf16)a.z, (__bf16)a.w };
        v4bf vb = { (__bf16)b.x, (__bf16)b.y, (__bf16)b.z, (__bf16)b.w };
        ((v4bf*)wih_bf)[i] = va;
        ((v4bf*)whh_bf)[i] = vb;
    }
}

// ---------------------------------------------------------------------------
// Kernel 2: x_proj = inputs @ W_ih^T + b_ih  -> d_out ([B*S, HID]).
// 1024 threads (32 waves), M-tile = 64 rows; each wave owns one 16-wide N
// slab x 4 M sub-tiles. B fragment loaded per k-block (used 4x, no array).
// ---------------------------------------------------------------------------
__global__ __launch_bounds__(1024) void rnn_xproj_gemm(
    const float*  __restrict__ inp,   // [B*S, IN] fp32
    const __bf16* __restrict__ wih,   // [HID, IN] bf16 row-major (n,k)
    const float*  __restrict__ bih,   // [HID]
    float*        __restrict__ out) { // [B*S, HID]
    __shared__ __bf16 As[MT * LDS_STRIDE];

    const int row0 = blockIdx.x * MT;

    // Stage 64x512 fp32 rows into LDS as bf16 (float4 in, 8B out)
    for (int idx = threadIdx.x; idx < MT * IN_DIM / 4; idx += 1024) {
        int m  = idx >> 7;          // 128 float4 per row
        int k4 = (idx & 127) * 4;
        float4 f = *(const float4*)(inp + (size_t)(row0 + m) * IN_DIM + k4);
        v4bf vb = { (__bf16)f.x, (__bf16)f.y, (__bf16)f.z, (__bf16)f.w };
        *(v4bf*)(As + m * LDS_STRIDE + k4) = vb;
    }
    __syncthreads();

    const int lane = threadIdx.x & 31;
    const int wv   = threadIdx.x >> 5;
    const int hf   = lane >> 4;
    const int mn   = lane & 15;
    const int nn   = wv * 16 + mn;   // this lane's N column

    const float bias = bih[nn];

    v8f acc[4];
#pragma unroll
    for (int ms = 0; ms < 4; ++ms)
#pragma unroll
        for (int r = 0; r < 8; ++r) acc[ms][r] = 0.0f;

    const __bf16* wrow = wih + (size_t)nn * IN_DIM;

#pragma unroll
    for (int kb = 0; kb < 16; ++kb) {
        const int kbase = kb * 32 + 8 * hf;
        Frag B;
        B.q[0] = *(const v4u*)(wrow + kbase);
        B.q[1] = *(const v4u*)(wrow + kbase + 16);
#pragma unroll
        for (int ms = 0; ms < 4; ++ms) {
            Frag A;
            const __bf16* arow = As + (ms * 16 + mn) * LDS_STRIDE;
            A.q[0] = *(const v4u*)(arow + kbase);
            A.q[1] = *(const v4u*)(arow + kbase + 16);
            acc[ms] = __builtin_amdgcn_wmma_f32_16x16x32_bf16(
                false, A.v, false, B.v, (short)0, acc[ms], false, false);
        }
    }

    // D layout: lane n = lane&15, vgpr r -> m = r + 8*(lane>>4)
#pragma unroll
    for (int ms = 0; ms < 4; ++ms) {
#pragma unroll
        for (int r = 0; r < 8; ++r) {
            int m = ms * 16 + r + hf * 8;
            out[(size_t)(row0 + m) * HID + nn] = acc[ms][r] + bias;
        }
    }
}

// ---------------------------------------------------------------------------
// Kernel 3: sequential scan. 4 blocks (16 batch rows each), 512 threads
// (16 waves -> 4 waves/SIMD -> large VGPR budget). Each wave owns TWO
// 16-wide N-tiles; W_hh fragments register-resident across all 2048 steps
// (Bf[2][16] = 256 VGPRs); h lives in LDS as bf16; x_proj read from d_out
// and overwritten in place by h_t.
// ---------------------------------------------------------------------------
__global__ __launch_bounds__(512) void rnn_scan(
    const __bf16* __restrict__ whh,     // [HID, HID] bf16 (n,k)
    const float*  __restrict__ h0,      // [B, HID]
    const float*  __restrict__ bhh,     // [HID]
    float*        __restrict__ out,     // [B, S, HID]: xp in, h out
    float*        __restrict__ hfinal) {// [B, HID]
    __shared__ __bf16 Hs[16 * LDS_STRIDE];

    const int b0 = blockIdx.x * 16;

    // Stage h0 into LDS as bf16
    for (int idx = threadIdx.x; idx < 16 * HID / 4; idx += 512) {
        int m  = idx >> 7;
        int k4 = (idx & 127) * 4;
        float4 f = *(const float4*)(h0 + (size_t)(b0 + m) * HID + k4);
        v4bf vb = { (__bf16)f.x, (__bf16)f.y, (__bf16)f.z, (__bf16)f.w };
        *(v4bf*)(Hs + m * LDS_STRIDE + k4) = vb;
    }

    const int lane = threadIdx.x & 31;
    const int wv   = threadIdx.x >> 5;
    const int hf   = lane >> 4;
    const int mn   = lane & 15;
    const int nnA  = wv * 32 + mn;        // first N-tile column
    const int nnB  = wv * 32 + 16 + mn;   // second N-tile column

    // Register-resident W_hh fragments for both N-tiles (reused 2048x)
    Frag Bf[2][16];
    {
        const __bf16* rowA = whh + (size_t)nnA * HID;
        const __bf16* rowB = whh + (size_t)nnB * HID;
#pragma unroll
        for (int kb = 0; kb < 16; ++kb) {
            const int kbase = kb * 32 + 8 * hf;
            Bf[0][kb].q[0] = *(const v4u*)(rowA + kbase);
            Bf[0][kb].q[1] = *(const v4u*)(rowA + kbase + 16);
            Bf[1][kb].q[0] = *(const v4u*)(rowB + kbase);
            Bf[1][kb].q[1] = *(const v4u*)(rowB + kbase + 16);
        }
    }
    const float biasA = bhh[nnA];
    const float biasB = bhh[nnB];
    __syncthreads();

    float hvA[8], hvB[8];
#pragma unroll 1
    for (int t = 0; t < SEQ; ++t) {
        v8f accA, accB;
        // accumulators start at x_proj_t + b_hh (read from d_out)
#pragma unroll
        for (int r = 0; r < 8; ++r) {
            int m = r + hf * 8;
            size_t rowoff = ((size_t)(b0 + m) * SEQ + t) * HID;
            accA[r] = out[rowoff + nnA] + biasA;
            accB[r] = out[rowoff + nnB] + biasB;
        }
        // hide next step's x_proj latency (global_prefetch_b8)
        if (t + 1 < SEQ) {
#pragma unroll
            for (int r = 0; r < 8; ++r) {
                int m = r + hf * 8;
                size_t rowoff = ((size_t)(b0 + m) * SEQ + (t + 1)) * HID;
                __builtin_prefetch(&out[rowoff + nnA], 0, 1);
            }
        }
        // h_{t-1} @ W_hh^T : one shared A fragment feeds both N-tiles
#pragma unroll
        for (int kb = 0; kb < 16; ++kb) {
            const int kbase = kb * 32 + 8 * hf;
            Frag A;
            const __bf16* arow = Hs + mn * LDS_STRIDE;
            A.q[0] = *(const v4u*)(arow + kbase);
            A.q[1] = *(const v4u*)(arow + kbase + 16);
            accA = __builtin_amdgcn_wmma_f32_16x16x32_bf16(
                false, A.v, false, Bf[0][kb].v, (short)0, accA, false, false);
            accB = __builtin_amdgcn_wmma_f32_16x16x32_bf16(
                false, A.v, false, Bf[1][kb].v, (short)0, accB, false, false);
        }
#pragma unroll
        for (int r = 0; r < 8; ++r) { hvA[r] = tanhf(accA[r]); hvB[r] = tanhf(accB[r]); }
        // write h_t (fp32) over x_proj_t in d_out
#pragma unroll
        for (int r = 0; r < 8; ++r) {
            int m = r + hf * 8;
            size_t rowoff = ((size_t)(b0 + m) * SEQ + t) * HID;
            out[rowoff + nnA] = hvA[r];
            out[rowoff + nnB] = hvB[r];
        }
        __syncthreads();   // all waves done reading old Hs
#pragma unroll
        for (int r = 0; r < 8; ++r) {
            int m = r + hf * 8;
            Hs[m * LDS_STRIDE + nnA] = (__bf16)hvA[r];
            Hs[m * LDS_STRIDE + nnB] = (__bf16)hvB[r];
        }
        __syncthreads();   // new Hs visible before next step
    }

#pragma unroll
    for (int r = 0; r < 8; ++r) {
        int m = r + hf * 8;
        hfinal[(size_t)(b0 + m) * HID + nnA] = hvA[r];
        hfinal[(size_t)(b0 + m) * HID + nnB] = hvB[r];
    }
}

// ---------------------------------------------------------------------------
extern "C" void kernel_launch(void* const* d_in, const int* in_sizes, int n_in,
                              void* d_out, int out_size, void* d_ws, size_t ws_size,
                              hipStream_t stream) {
    const float* inputs = (const float*)d_in[0];  // [B,S,IN]
    const float* h0     = (const float*)d_in[1];  // [B,HID]
    const float* W_ih   = (const float*)d_in[2];  // [HID,IN]
    const float* W_hh   = (const float*)d_in[3];  // [HID,HID]
    const float* b_ih   = (const float*)d_in[4];  // [HID]
    const float* b_hh   = (const float*)d_in[5];  // [HID]

    float* out    = (float*)d_out;
    float* hfinal = out + (size_t)BATCH * SEQ * HID;

    __bf16* wih_bf = (__bf16*)d_ws;                       // 512 KB
    __bf16* whh_bf = wih_bf + (size_t)HID * IN_DIM;       // 512 KB

    rnn_cvt_weights<<<(HID * IN_DIM / 4 + 255) / 256, 256, 0, stream>>>(
        W_ih, W_hh, wih_bf, whh_bf);

    rnn_xproj_gemm<<<(BATCH * SEQ) / MT, 1024, 0, stream>>>(
        inputs, wih_bf, b_ih, out);

    rnn_scan<<<BATCH / 16, 512, 0, stream>>>(
        whh_bf, h0, b_hh, out, hfinal);
}
